// MambaLayer_4612794876629
// MI455X (gfx1250) — compile-verified, matrix operand order checked by reading, not run on hardware
//
#include <hip/hip_runtime.h>
#include <math.h>
#include <stdint.h>

// ---------------- problem constants ----------------
#define LSEQ     4096
#define BATCH    4
#define DMODEL   1024
#define DINNER   2048
#define DSTATE   16
#define DCONV    4
#define DTRANK   64
#define MROWS    (BATCH * LSEQ)        // 16384

typedef unsigned short ushort_t;

// ---------------- WMMA types ----------------
typedef __attribute__((ext_vector_type(16))) __bf16 v16bf;
typedef __attribute__((ext_vector_type(8)))  float  v8f;

union BF16x16 { uint4 q[2]; v16bf v; };

__device__ __forceinline__ ushort_t f2bf(float f) {
  unsigned int u = __float_as_uint(f);
  u += 0x7FFFu + ((u >> 16) & 1u);      // round-to-nearest-even
  return (ushort_t)(u >> 16);
}
__device__ __forceinline__ float bf2f(ushort_t u) {
  return __uint_as_float(((unsigned int)u) << 16);
}

// ---------------- CDNA5 async global->LDS copy ----------------
#if defined(__has_builtin)
#if __has_builtin(__builtin_amdgcn_global_load_async_to_lds_b128) && \
    __has_builtin(__builtin_amdgcn_s_wait_asynccnt)
#define ASYNC_OK 1
#endif
#endif

// builtin wants: (v4i addrspace(1)* gsrc, v4i addrspace(3)* ldst, imm, imm)
typedef int v4i_ __attribute__((vector_size(16)));
typedef __attribute__((address_space(1))) v4i_ GV4;   // global 128-bit chunk
typedef __attribute__((address_space(3))) v4i_ LV4;   // LDS 128-bit chunk

__device__ __forceinline__ void async_copy16(void* lds_dst, const void* gsrc) {
#ifdef ASYNC_OK
  __builtin_amdgcn_global_load_async_to_lds_b128(
      (GV4*)(unsigned long long)(uintptr_t)gsrc,
      (LV4*)(unsigned int)(uintptr_t)lds_dst,
      0, 0);
#else
  *reinterpret_cast<uint4*>(lds_dst) = *reinterpret_cast<const uint4*>(gsrc);
#endif
}

#ifdef ASYNC_OK
#define WAIT_ASYNC(n) __builtin_amdgcn_s_wait_asynccnt(n)
#else
#define WAIT_ASYNC(n)
#endif

// =====================================================================
// f32 -> bf16 conversion kernels (run once per launch).
// =====================================================================
__global__ __launch_bounds__(256)
void cvt_bf16_kernel(const float* __restrict__ src, ushort_t* __restrict__ dst) {
  int i = (blockIdx.x * 256 + threadIdx.x) * 4;
  float4 v = *reinterpret_cast<const float4*>(src + i);
  dst[i + 0] = f2bf(v.x); dst[i + 1] = f2bf(v.y);
  dst[i + 2] = f2bf(v.z); dst[i + 3] = f2bf(v.w);
}

// x: (L,B,D) f32 -> xbf: (B*L, D) bf16 with row remap m=b*L+l <- l*B+b
__global__ __launch_bounds__(256)
void x_remap_bf16_kernel(const float* __restrict__ x, ushort_t* __restrict__ xbf) {
  int idx = blockIdx.x * 256 + threadIdx.x;   // one float4 each
  int m  = idx >> 8;                          // 256 float4 per row
  int c4 = (idx & 255) << 2;
  int b = m / LSEQ, l = m % LSEQ;
  float4 v = *reinterpret_cast<const float4*>(x + ((size_t)l * BATCH + b) * DMODEL + c4);
  ushort_t* d = xbf + (size_t)m * DMODEL + c4;
  d[0] = f2bf(v.x); d[1] = f2bf(v.y); d[2] = f2bf(v.z); d[3] = f2bf(v.w);
}

// =====================================================================
// Unified NT GEMM, all-bf16 operands:  C[M,N] = A[M,K] * W[N,K]^T
// bf16 V_WMMA_F32_16X16X32_BF16, f32 accumulate.
// Double-buffered LDS tiles filled with GLOBAL_LOAD_ASYNC_TO_LDS_B128.
//   EPI==1 : C = softplus(C + bias[n])
//   OMAP   : logical row m=b*L+l stored at physical row l*B+b
//   OBF16  : C stored as bf16 (else f32)
// Tile 128x128x64, 256 threads = 8 waves, wave tile 64x32 (4x2 frags).
// =====================================================================
template<int EPI, bool OMAP, bool OBF16>
__global__ __launch_bounds__(256)
void gemm_wmma_bf16(const ushort_t* __restrict__ A, int lda,
                    const ushort_t* __restrict__ W, int ldw,
                    void* __restrict__ Cp, int ldc,
                    int M, int N, int K,
                    const float* __restrict__ bias)
{
  constexpr int BM = 128, BN = 128, BK = 64;
  __shared__ alignas(16) ushort_t As[2][BM][BK];   // 2 x 16 KB
  __shared__ alignas(16) ushort_t Bs[2][BN][BK];   // 2 x 16 KB

  const int tid  = threadIdx.x;
  const int lane = tid & 31;
  const int wid  = tid >> 5;
  const int wm   = wid >> 2;          // 0..1  (64 rows)
  const int wn   = wid & 3;           // 0..3  (32 cols)
  const int bm0  = blockIdx.y * BM;
  const int bn0  = blockIdx.x * BN;

  v8f acc[4][2];
  #pragma unroll
  for (int i = 0; i < 4; ++i)
    #pragma unroll
    for (int j = 0; j < 2; ++j) acc[i][j] = {};

  // ---- async tile stage: 8 x b128 per wave (4 A + 4 B) ----
  auto stage = [&](int s, int k0) {
    #pragma unroll
    for (int i = 0; i < 4; ++i) {
      int idx = tid + i * 256;          // 1024 chunks of 8 bf16
      int row = idx >> 3;
      int c8  = (idx & 7) << 3;
      int m = bm0 + row; if (m >= M) m = M - 1;   // clamp: junk cols/rows
      int n = bn0 + row; if (n >= N) n = N - 1;   // are never stored
      async_copy16(&As[s][row][c8], A + (size_t)m * lda + k0 + c8);
      async_copy16(&Bs[s][row][c8], W + (size_t)n * ldw + k0 + c8);
    }
  };

  const int nk = K / BK;
  stage(0, 0);

  for (int kt = 0; kt < nk; ++kt) {
    const int s = kt & 1;
    if (kt + 1 < nk) {
      stage(s ^ 1, (kt + 1) * BK);
      WAIT_ASYNC(8);     // tile kt landed; next tile (8 asyncs) may fly
    } else {
      WAIT_ASYNC(0);
    }
    __syncthreads();

    #pragma unroll
    for (int k2 = 0; k2 < BK / 32; ++k2) {
      // 16-bit fragment K pattern (ISA 7.12.2):
      // elem e -> K = base + (e&7) + 16*(e>>3), base = (lane>=16)?8:0
      const int ko = k2 * 32 + ((lane >> 4) << 3);
      v16bf afrag[4], bfrag[2];
      #pragma unroll
      for (int mf = 0; mf < 4; ++mf) {
        int row = wm * 64 + mf * 16 + (lane & 15);
        BF16x16 u;
        u.q[0] = *reinterpret_cast<const uint4*>(&As[s][row][ko]);
        u.q[1] = *reinterpret_cast<const uint4*>(&As[s][row][ko + 16]);
        afrag[mf] = u.v;
      }
      #pragma unroll
      for (int nf = 0; nf < 2; ++nf) {
        int row = wn * 32 + nf * 16 + (lane & 15);
        BF16x16 u;
        u.q[0] = *reinterpret_cast<const uint4*>(&Bs[s][row][ko]);
        u.q[1] = *reinterpret_cast<const uint4*>(&Bs[s][row][ko + 16]);
        bfrag[nf] = u.v;
      }
      #pragma unroll
      for (int mf = 0; mf < 4; ++mf)
        #pragma unroll
        for (int nf = 0; nf < 2; ++nf)
          acc[mf][nf] = __builtin_amdgcn_wmma_f32_16x16x32_bf16(
              false, afrag[mf], false, bfrag[nf],
              (short)0, acc[mf][nf], false, false);
    }
    __syncthreads();
  }

  // ---- epilogue: D layout = VGPR r -> M = r + 8*(lane>=16), N = lane&15
  const int mi   = (lane >> 4) << 3;
  const int ncol = lane & 15;
  #pragma unroll
  for (int mf = 0; mf < 4; ++mf) {
    #pragma unroll
    for (int nf = 0; nf < 2; ++nf) {
      #pragma unroll
      for (int r = 0; r < 8; ++r) {
        int m = bm0 + wm * 64 + mf * 16 + r + mi;
        int n = bn0 + wn * 32 + nf * 16 + ncol;
        if (m < M && n < N) {
          float v = acc[mf][nf][r];
          if constexpr (EPI == 1) {
            v += bias[n];
            v = (v > 20.f) ? v : log1pf(__expf(v));   // softplus
          }
          int crow = m;
          if constexpr (OMAP) { int b = m / LSEQ, l = m % LSEQ; crow = l * BATCH + b; }
          if constexpr (OBF16)
            ((ushort_t*)Cp)[(size_t)crow * ldc + n] = f2bf(v);
          else
            ((float*)Cp)[(size_t)crow * ldc + n] = v;
        }
      }
    }
  }
}

// =====================================================================
// Depthwise causal conv (width 4) + SiLU, bf16 in/out (f32 math).
// =====================================================================
__global__ __launch_bounds__(256)
void conv_silu_kernel(const ushort_t* __restrict__ xz,
                      const float* __restrict__ cw,
                      const float* __restrict__ cb,
                      ushort_t* __restrict__ xc)
{
  int tid = blockIdx.x * blockDim.x + threadIdx.x;
  int d  = tid & (DINNER - 1);
  int lt = (tid >> 11) & (LSEQ / 8 - 1);
  int b  = tid >> 20;
  if (b >= BATCH) return;

  float w0 = cw[d * DCONV + 0], w1 = cw[d * DCONV + 1];
  float w2 = cw[d * DCONV + 2], w3 = cw[d * DCONV + 3];
  float bv = cb[d];
  int l0 = lt * 8;

  float win[11];
  #pragma unroll
  for (int j = 0; j < 11; ++j) {
    int l = l0 - 3 + j;
    win[j] = (l >= 0) ? bf2f(xz[(size_t)(b * LSEQ + l) * (2 * DINNER) + d]) : 0.f;
  }
  #pragma unroll
  for (int o = 0; o < 8; ++o) {
    float s = bv + w0 * win[o] + w1 * win[o + 1] + w2 * win[o + 2] + w3 * win[o + 3];
    float y = s / (1.f + __expf(-s));        // SiLU
    xc[(size_t)(b * LSEQ + l0 + o) * DINNER + d] = f2bf(y);
  }
}

// =====================================================================
// Selective scan: 16 lanes/channel (one per state), 16 channels per WG.
// LDS chunks staged with async global->LDS copies.
// =====================================================================
__global__ __launch_bounds__(256)
void scan_kernel(const ushort_t* __restrict__ xz,     // z gate cols [2048,4096)
                 const ushort_t* __restrict__ xc,
                 const ushort_t* __restrict__ xdbl,   // [.,96]: dt_lo | B | C
                 const float* __restrict__ dt,
                 const float* __restrict__ A_log,
                 const float* __restrict__ Dp,
                 ushort_t* __restrict__ y)
{
  constexpr int CL = 64;
  __shared__ alignas(16) ushort_t sB [CL][DSTATE];
  __shared__ alignas(16) ushort_t sC [CL][DSTATE];
  __shared__ alignas(16) ushort_t sx [CL][16];
  __shared__ alignas(16) ushort_t sz [CL][16];
  __shared__ alignas(16) float    sdt[CL][16];

  const int tid  = threadIdx.x;
  const int lane = tid & 31;
  const int s    = lane & 15;                       // state index
  const int ch   = ((tid >> 5) << 1) | (lane >> 4); // 0..15 channel-in-group
  const int b    = blockIdx.x >> 7;                 // DINNER/16 = 128 groups
  const int dg   = blockIdx.x & 127;
  const int d0   = dg * 16;
  const int d    = d0 + ch;

  const float Asv = -__expf(A_log[d * DSTATE + s]);
  const float Dv  = Dp[d];

  float h = 0.f;
  for (int l0 = 0; l0 < LSEQ; l0 += CL) {
    // ---- async stage: bf16 rows are 32B (2 x b128), dt rows 64B (4 x b128)
    if (tid < 128) {
      int li = tid >> 1, p = (tid & 1) * 8;
      size_t row = (size_t)(b * LSEQ + l0 + li);
      const ushort_t* xb = xdbl + row * 96;
      async_copy16(&sB[li][p], xb + DTRANK + p);
      async_copy16(&sC[li][p], xb + DTRANK + DSTATE + p);
      async_copy16(&sx[li][p], xc + row * DINNER + d0 + p);
      async_copy16(&sz[li][p], xz + row * (2 * DINNER) + DINNER + d0 + p);
    }
    {
      int li = tid >> 2, p = (tid & 3) * 4;
      size_t row = (size_t)(b * LSEQ + l0 + li);
      async_copy16(&sdt[li][p], dt + row * DINNER + d0 + p);
    }
    WAIT_ASYNC(0);
    __syncthreads();

    for (int li = 0; li < CL; ++li) {
      float dtv = sdt[li][ch];
      float xv  = bf2f(sx[li][ch]);
      float dA  = __expf(dtv * Asv);
      h = dA * h + dtv * bf2f(sB[li][s]) * xv;
      float p = h * bf2f(sC[li][s]);
      p += __shfl_xor(p, 1, 32);
      p += __shfl_xor(p, 2, 32);
      p += __shfl_xor(p, 4, 32);
      p += __shfl_xor(p, 8, 32);                    // sum over 16 states
      if (s == 0) {
        float yv = p + Dv * xv;
        float zv = bf2f(sz[li][ch]);
        y[(size_t)(b * LSEQ + l0 + li) * DINNER + d] =
            f2bf(yv * (zv / (1.f + __expf(-zv))));
      }
    }
    __syncthreads();
  }
}

// =====================================================================
extern "C" void kernel_launch(void* const* d_in, const int* in_sizes, int n_in,
                              void* d_out, int out_size, void* d_ws, size_t ws_size,
                              hipStream_t stream) {
  const float* x      = (const float*)d_in[0];
  const float* W_in   = (const float*)d_in[1];
  const float* conv_w = (const float*)d_in[2];
  const float* conv_b = (const float*)d_in[3];
  const float* W_xp   = (const float*)d_in[4];
  const float* W_dt   = (const float*)d_in[5];
  const float* b_dt   = (const float*)d_in[6];
  const float* A_log  = (const float*)d_in[7];
  const float* Dp     = (const float*)d_in[8];
  const float* W_out  = (const float*)d_in[9];
  float* out = (float*)d_out;

  // ---- workspace layout (bf16 intermediates) ----
  char* ws = (char*)d_ws;
  const size_t SZ_XBF  = (size_t)MROWS * DMODEL * 2;       // 32 MB
  const size_t SZ_WIN  = (size_t)2 * DINNER * DMODEL * 2;  // 8 MB
  const size_t SZ_WXP  = (size_t)96 * DINNER * 2;
  const size_t SZ_WDT  = (size_t)DINNER * DTRANK * 2;
  const size_t SZ_WOUT = (size_t)DMODEL * DINNER * 2;
  const size_t SZ_XZ   = (size_t)MROWS * 2 * DINNER * 2;   // 128 MB
  const size_t SZ_XC   = (size_t)MROWS * DINNER * 2;       // 64 MB
  const size_t SZ_XDBL = (size_t)MROWS * 96 * 2;
  const size_t SZ_DT   = (size_t)MROWS * DINNER * 4;       // 128 MB

  size_t off = 0;
  ushort_t* xbf   = (ushort_t*)(ws + off); off += SZ_XBF;
  ushort_t* wInb  = (ushort_t*)(ws + off); off += SZ_WIN;
  ushort_t* wXpb  = (ushort_t*)(ws + off); off += SZ_WXP;
  ushort_t* wDtb  = (ushort_t*)(ws + off); off += SZ_WDT;
  ushort_t* wOutb = (ushort_t*)(ws + off); off += SZ_WOUT;
  ushort_t* xz    = (ushort_t*)(ws + off); off += SZ_XZ;
  ushort_t* xc    = (ushort_t*)(ws + off); off += SZ_XC;
  ushort_t* xdbl  = (ushort_t*)(ws + off); off += SZ_XDBL;
  float*    dtb   = (float*)   (ws + off); off += SZ_DT;
  ushort_t* yb    = (ushort_t*)(ws + off);

  dim3 blk(256);

  // 0) one-time f32 -> bf16 materialization (weights + remapped input)
  x_remap_bf16_kernel<<<(MROWS * DMODEL / 4) / 256, blk, 0, stream>>>(x, xbf);
  cvt_bf16_kernel<<<(2 * DINNER * DMODEL) / 1024, blk, 0, stream>>>(W_in, wInb);
  cvt_bf16_kernel<<<(96 * DINNER) / 1024,          blk, 0, stream>>>(W_xp, wXpb);
  cvt_bf16_kernel<<<(DINNER * DTRANK) / 1024,      blk, 0, stream>>>(W_dt, wDtb);
  cvt_bf16_kernel<<<(DMODEL * DINNER) / 1024,      blk, 0, stream>>>(W_out, wOutb);

  // 1) xz = xb @ W_in^T   (M=16384, N=4096, K=1024) -> bf16
  gemm_wmma_bf16<0, false, true>
      <<<dim3(2 * DINNER / 128, MROWS / 128), blk, 0, stream>>>(
          xbf, DMODEL, wInb, DMODEL, xz, 2 * DINNER,
          MROWS, 2 * DINNER, DMODEL, nullptr);

  // 2) depthwise conv + SiLU
  conv_silu_kernel<<<(BATCH * DINNER * (LSEQ / 8)) / 256, blk, 0, stream>>>(
      xz, conv_w, conv_b, xc);

  // 3) xdbl = xc @ W_xproj^T   (N=96, K=2048) -> bf16
  gemm_wmma_bf16<0, false, true>
      <<<dim3(1, MROWS / 128), blk, 0, stream>>>(
          xc, DINNER, wXpb, DINNER, xdbl, 96,
          MROWS, 96, DINNER, nullptr);

  // 4) dt = softplus(dt_lo @ W_dt^T + b_dt)   (N=2048, K=64) -> f32
  gemm_wmma_bf16<1, false, false>
      <<<dim3(DINNER / 128, MROWS / 128), blk, 0, stream>>>(
          xdbl, 96, wDtb, DTRANK, dtb, DINNER,
          MROWS, DINNER, DTRANK, b_dt);

  // 5) selective scan + skip + SiLU gate -> bf16
  scan_kernel<<<BATCH * (DINNER / 16), blk, 0, stream>>>(
      xz, xc, xdbl, dtb, A_log, Dp, yb);

  // 6) out = y @ W_out^T, rows remapped back to (L, B, D_MODEL) -> f32
  gemm_wmma_bf16<0, true, false>
      <<<dim3(DMODEL / 128, MROWS / 128), blk, 0, stream>>>(
          yb, DINNER, wOutb, DINNER, out, DMODEL,
          MROWS, DMODEL, DINNER, nullptr);
}
